// TwoHopGATBlock_83459804496490
// MI455X (gfx1250) — matrix operand order repaired and checked by reading, not compiled, not run on hardware
//
#include <hip/hip_runtime.h>

#define NB   16
#define NN   1024
#define NE   32768
#define DIN  33
#define HID  32
#define NH   4
#define DOUT 8
#define NEG_SLOPE 0.2f
#define EPS 1e-5f

typedef __attribute__((ext_vector_type(16))) _Float16 v16h;
typedef __attribute__((ext_vector_type(8)))  float    v8f;
typedef __attribute__((ext_vector_type(4)))  float    f4;

union HF16 { v16h v; _Float16 h[16]; };
union F8   { v8f  v; float    f[8];  };

// ---------------- WMMA helpers (gfx1250 layouts, wave32) ----------------

// A-matrix 16x32 f16 fragment from a row-major f32 tile (lane<16: M=lane).
// element i -> K = i + 16*half. Caller guarantees lda/kmax
// such that all accesses are in range (guards fold to true).
__device__ inline v16h frag_a_f32(const float* __restrict__ src, int lda,
                                  int kmax, int k0, int lane) {
  const int half = (lane >> 4) & 1;
  const int r = lane & 15;
  HF16 u;
#pragma unroll
  for (int i = 0; i < 16; ++i) {
    int K = i + 16 * half;
    int kk = k0 + K;
    float x = (kk < kmax) ? src[(size_t)r * lda + kk] : 0.0f;
    u.h[i] = (_Float16)x;
  }
  return u.v;
}

// B-matrix 32x16 f16 fragment from row-major f32 weights [K x ldw], cols c0..c0+15.
// element i -> K = i + 16*half ; lane -> column
__device__ inline v16h frag_b_f32(const float* __restrict__ W, int ldw,
                                  int kmax, int k0, int c0, int lane) {
  const int half = (lane >> 4) & 1;
  const int r = lane & 15;
  HF16 u;
#pragma unroll
  for (int i = 0; i < 16; ++i) {
    int kk = k0 + i + 16 * half;
    float x = (kk < kmax) ? W[(size_t)kk * ldw + c0 + r] : 0.0f;
    u.h[i] = (_Float16)x;
  }
  return u.v;
}

__device__ inline v8f wmma16(v16h a, v16h b, v8f c) {
  // D = A*B + C, fp32 accumulate. 8-arg probe-confirmed signature.
  return __builtin_amdgcn_wmma_f32_16x16x32_f16(false, a, false, b,
                                                (short)0, c, false, false);
}

// order-preserving float<->uint map for atomicMax on floats
__device__ inline unsigned f2ord(float f) {
  unsigned u = __float_as_uint(f);
  return (u >> 31) ? ~u : (u | 0x80000000u);
}
__device__ inline float ord2f(unsigned o) {
  unsigned u = (o >> 31) ? (o & 0x7FFFFFFFu) : ~o;
  return __uint_as_float(u);
}

// ---------------- Kernels ----------------

// Streaming zero of the dense attention output A (268 MB) — the HBM-bound part.
__global__ __launch_bounds__(256) void zero_a_kernel(f4* __restrict__ p, size_t n4) {
  size_t stride = (size_t)gridDim.x * blockDim.x;
  f4 z = (f4){0.f, 0.f, 0.f, 0.f};
  for (size_t i = (size_t)blockIdx.x * blockDim.x + threadIdx.x; i < n4; i += stride)
    __builtin_nontemporal_store(z, &p[i]);
}

// Re-init of atomic scratch (segment max / sum / aggregation) each call.
__global__ __launch_bounds__(256) void init_ws_kernel(unsigned* __restrict__ M,
                                                      float* __restrict__ S,
                                                      float* __restrict__ AGG) {
  int idx = blockIdx.x * blockDim.x + threadIdx.x;          // 524288 threads
  AGG[idx] = 0.f;
  if (idx < NB * NN * NH) { M[idx] = 0x007FFFFFu; S[idx] = 0.f; }  // ord(-inf)
}

// ObsEmbedding: relu(H@W1+b1) -> relu(@W2+b2) -> LayerNorm. 128 rows/block, WMMA.
// K=33 GEMM is zero-padded in LDS so fragment loads are branch-free and EXEC
// stays all-1s around every WMMA.
__global__ __launch_bounds__(256) void embed_kernel(
    const float* __restrict__ H, const float* __restrict__ W1, const float* __restrict__ b1,
    const float* __restrict__ W2, const float* __restrict__ b2,
    const float* __restrict__ lg, const float* __restrict__ lb, float* __restrict__ X) {
  __shared__ float htile[8][16][64];   // per-wave H rows, K zero-padded 33->64
  __shared__ float w1s[64][32];        // W1 zero-padded 33->64 rows
  __shared__ float tile[8][16][33];    // intermediate activations
  const int lane = threadIdx.x & 31;
  const int w = threadIdx.x >> 5;
  const int half = (lane >> 4) & 1;
  const int r = lane & 15;
  const int row0 = blockIdx.x * 128 + w * 16;

  // Stage W1 (block-cooperative, zero-padded)
  for (int i = threadIdx.x; i < 64 * 32; i += 256) {
    int k = i >> 5, c = i & 31;
    w1s[k][c] = (k < DIN) ? W1[k * 32 + c] : 0.f;
  }
  // Stage this wave's 16x33 H rows (coalesced, zero-padded to 64)
  for (int i = lane; i < 16 * 64; i += 32) {
    int rr = i >> 6, cc = i & 63;
    htile[w][rr][cc] = (cc < DIN) ? H[(size_t)(row0 + rr) * DIN + cc] : 0.f;
  }
  __syncthreads();

  // GEMM1: [16 x 64pad] @ [64pad x 32]
  F8 c0, c1;
#pragma unroll
  for (int q = 0; q < 8; ++q) { c0.f[q] = 0.f; c1.f[q] = 0.f; }
#pragma unroll
  for (int k0 = 0; k0 < 64; k0 += 32) {
    v16h a  = frag_a_f32(&htile[w][0][0], 64, 64, k0, lane);
    v16h bl = frag_b_f32(&w1s[0][0], HID, 64, k0, 0, lane);
    v16h br = frag_b_f32(&w1s[0][0], HID, 64, k0, 16, lane);
    c0.v = wmma16(a, bl, c0.v);
    c1.v = wmma16(a, br, c1.v);
  }
#pragma unroll
  for (int q = 0; q < 8; ++q) {
    int m = q + 8 * half;
    tile[w][m][r]      = fmaxf(c0.f[q] + b1[r], 0.f);
    tile[w][m][r + 16] = fmaxf(c1.f[q] + b1[r + 16], 0.f);
  }
  __syncthreads();

  // GEMM2: [16 x 32] @ [32 x 32] from LDS (kmax==32: guards fold away)
  v16h a2  = frag_a_f32(&tile[w][0][0], 33, HID, 0, lane);
  v16h bl2 = frag_b_f32(W2, HID, HID, 0, 0, lane);
  v16h br2 = frag_b_f32(W2, HID, HID, 0, 16, lane);
  F8 d0, d1;
#pragma unroll
  for (int q = 0; q < 8; ++q) { d0.f[q] = 0.f; d1.f[q] = 0.f; }
  d0.v = wmma16(a2, bl2, d0.v);
  d1.v = wmma16(a2, br2, d1.v);
  __syncthreads();
#pragma unroll
  for (int q = 0; q < 8; ++q) {
    int m = q + 8 * half;
    tile[w][m][r]      = fmaxf(d0.f[q] + b2[r], 0.f);
    tile[w][m][r + 16] = fmaxf(d1.f[q] + b2[r + 16], 0.f);
  }
  __syncthreads();

  // LayerNorm per row, one thread per row
  if (threadIdx.x < 128) {
    const float* rowp = &tile[threadIdx.x >> 4][threadIdx.x & 15][0];
    float mu = 0.f;
#pragma unroll
    for (int j = 0; j < 32; ++j) mu += rowp[j];
    mu *= (1.f / 32.f);
    float var = 0.f;
#pragma unroll
    for (int j = 0; j < 32; ++j) { float dd = rowp[j] - mu; var += dd * dd; }
    var *= (1.f / 32.f);
    float inv = rsqrtf(var + EPS);
    size_t gr = (size_t)blockIdx.x * 128 + threadIdx.x;
#pragma unroll
    for (int j = 0; j < 32; ++j)
      X[gr * 32 + j] = (rowp[j] - mu) * inv * lg[j] + lb[j];
  }
}

// xp = Xin @ W (shared across batches) plus per-head attention logits.
__global__ __launch_bounds__(256) void xp_kernel(
    const float* __restrict__ Xin, const float* __restrict__ W,
    const float* __restrict__ asrc, const float* __restrict__ adst,
    float* __restrict__ XP, float* __restrict__ ALS, float* __restrict__ ALD) {
  __shared__ float tile[8][16][33];
  const int lane = threadIdx.x & 31;
  const int w = threadIdx.x >> 5;
  const int half = (lane >> 4) & 1;
  const int r = lane & 15;
  const int row0 = blockIdx.x * 128 + w * 16;

  v16h a  = frag_a_f32(Xin + (size_t)row0 * HID, HID, HID, 0, lane);
  v16h bl = frag_b_f32(W, HID, HID, 0, 0, lane);
  v16h br = frag_b_f32(W, HID, HID, 0, 16, lane);
  F8 c0, c1;
#pragma unroll
  for (int q = 0; q < 8; ++q) { c0.f[q] = 0.f; c1.f[q] = 0.f; }
  c0.v = wmma16(a, bl, c0.v);
  c1.v = wmma16(a, br, c1.v);
#pragma unroll
  for (int q = 0; q < 8; ++q) {
    int m = q + 8 * half;
    tile[w][m][r]      = c0.f[q];
    tile[w][m][r + 16] = c1.f[q];
  }
  __syncthreads();
  if (threadIdx.x < 128) {
    const float* rowp = &tile[threadIdx.x >> 4][threadIdx.x & 15][0];
    size_t gr = (size_t)blockIdx.x * 128 + threadIdx.x;
#pragma unroll
    for (int j = 0; j < 32; ++j) XP[gr * 32 + j] = rowp[j];
#pragma unroll
    for (int h = 0; h < NH; ++h) {
      float s = 0.f, d = 0.f;
#pragma unroll
      for (int q = 0; q < DOUT; ++q) {
        float xv = rowp[h * DOUT + q];
        s += xv * asrc[h * DOUT + q];
        d += xv * adst[h * DOUT + q];
      }
      ALS[gr * NH + h] = s;
      ALD[gr * NH + h] = d;
    }
  }
}

// segment max over dst via ordered-uint atomicMax
__global__ __launch_bounds__(256) void edge_max_kernel(
    const long long* __restrict__ EI, const float* __restrict__ ALS,
    const float* __restrict__ ALD, unsigned* __restrict__ M) {
  int idx = blockIdx.x * blockDim.x + threadIdx.x;   // NB*NE
  int b = idx >> 15, e = idx & (NE - 1);
  int s = (int)EI[e], d = (int)EI[NE + e];
  const float* ps = ALS + ((size_t)b * NN + s) * NH;
  const float* pd = ALD + ((size_t)b * NN + d) * NH;
  unsigned* pm = M + ((size_t)b * NN + d) * NH;
#pragma unroll
  for (int h = 0; h < NH; ++h) {
    float v = ps[h] + pd[h];
    v = (v > 0.f) ? v : NEG_SLOPE * v;
    atomicMax(&pm[h], f2ord(v));
  }
}

// segment sum of exp(e - max)
__global__ __launch_bounds__(256) void edge_sum_kernel(
    const long long* __restrict__ EI, const float* __restrict__ ALS,
    const float* __restrict__ ALD, const unsigned* __restrict__ M,
    float* __restrict__ S) {
  int idx = blockIdx.x * blockDim.x + threadIdx.x;   // NB*NE
  int b = idx >> 15, e = idx & (NE - 1);
  int s = (int)EI[e], d = (int)EI[NE + e];
  const float* ps = ALS + ((size_t)b * NN + s) * NH;
  const float* pd = ALD + ((size_t)b * NN + d) * NH;
  const unsigned* pm = M + ((size_t)b * NN + d) * NH;
  float* pS = S + ((size_t)b * NN + d) * NH;
#pragma unroll
  for (int h = 0; h < NH; ++h) {
    float v = ps[h] + pd[h];
    v = (v > 0.f) ? v : NEG_SLOPE * v;
    atomicAdd(&pS[h], __expf(v - ord2f(pm[h])));
  }
}

// alpha = exp(e-max)/sum ; AGG[dst] += xp[src]*alpha ; layer2: scatter alpha into A
__global__ __launch_bounds__(256) void aggregate_kernel(
    const long long* __restrict__ EI, const float* __restrict__ ALS,
    const float* __restrict__ ALD, const unsigned* __restrict__ M,
    const float* __restrict__ S, const float* __restrict__ XP,
    float* __restrict__ AGG, float* __restrict__ A, int writeA) {
  int idx = blockIdx.x * blockDim.x + threadIdx.x;   // NB*NE*NH
  int h = idx & 3;
  int e = (idx >> 2) & (NE - 1);
  int b = idx >> 17;
  int s = (int)EI[e], d = (int)EI[NE + e];
  size_t ns = (size_t)b * NN + s, nd = (size_t)b * NN + d;
  const float* xs = XP + ns * HID + h * DOUT;
  __builtin_prefetch(xs, 0, 0);   // global_prefetch_b8: hide random gather latency
  float v = ALS[ns * NH + h] + ALD[nd * NH + h];
  v = (v > 0.f) ? v : NEG_SLOPE * v;
  float ex = __expf(v - ord2f(M[nd * NH + h]));
  float alpha = ex / S[nd * NH + h];
  if (writeA)
    A[(((size_t)b * NH + h) * NN + s) * NN + d] = alpha;
  float* ag = AGG + nd * HID + h * DOUT;
#pragma unroll
  for (int q = 0; q < DOUT; ++q) atomicAdd(&ag[q], xs[q] * alpha);
}

// h1 = relu(agg + bias)
__global__ __launch_bounds__(256) void post1_kernel(
    const float* __restrict__ AGG, const float* __restrict__ bi, float* __restrict__ H1) {
  int idx = blockIdx.x * blockDim.x + threadIdx.x;   // NB*NN*HID
  H1[idx] = fmaxf(AGG[idx] + bi[idx & 31], 0.f);
}

// h_out = LayerNorm(h1 + (agg2 + bias2))
__global__ __launch_bounds__(256) void post2_kernel(
    const float* __restrict__ H1, const float* __restrict__ AGG,
    const float* __restrict__ bi, const float* __restrict__ lg,
    const float* __restrict__ lb, float* __restrict__ OUT) {
  size_t gr = (size_t)blockIdx.x * blockDim.x + threadIdx.x;  // NB*NN rows
  float y[32];
  float mu = 0.f;
#pragma unroll
  for (int j = 0; j < 32; ++j) {
    y[j] = H1[gr * 32 + j] + AGG[gr * 32 + j] + bi[j];
    mu += y[j];
  }
  mu *= (1.f / 32.f);
  float var = 0.f;
#pragma unroll
  for (int j = 0; j < 32; ++j) { float dd = y[j] - mu; var += dd * dd; }
  var *= (1.f / 32.f);
  float inv = rsqrtf(var + EPS);
#pragma unroll
  for (int j = 0; j < 32; ++j)
    OUT[gr * 32 + j] = (y[j] - mu) * inv * lg[j] + lb[j];
}

// ---------------- Launch ----------------

extern "C" void kernel_launch(void* const* d_in, const int* in_sizes, int n_in,
                              void* d_out, int out_size, void* d_ws, size_t ws_size,
                              hipStream_t stream) {
  const float*     Hin = (const float*)d_in[0];
  const long long* EI  = (const long long*)d_in[1];   // jnp.int64
  const float* W1e = (const float*)d_in[2];
  const float* b1e = (const float*)d_in[3];
  const float* W2e = (const float*)d_in[4];
  const float* b2e = (const float*)d_in[5];
  const float* lneg = (const float*)d_in[6];
  const float* lneb = (const float*)d_in[7];
  const float* Wg1 = (const float*)d_in[8];
  const float* as1 = (const float*)d_in[9];
  const float* ad1 = (const float*)d_in[10];
  const float* bi1 = (const float*)d_in[11];
  const float* Wg2 = (const float*)d_in[12];
  const float* as2 = (const float*)d_in[13];
  const float* ad2 = (const float*)d_in[14];
  const float* bi2 = (const float*)d_in[15];
  const float* lnog = (const float*)d_in[16];
  const float* lnob = (const float*)d_in[17];

  float* outf = (float*)d_out;
  float* Hout = outf;                                  // [B,N,32]
  float* Amat = outf + (size_t)NB * NN * HID;          // [B,H,N,N]

  // workspace layout (f32 units)
  float* ws = (float*)d_ws;
  float* X    = ws;                                    // 524288
  float* XP   = X + (size_t)NB * NN * HID;             // 524288
  float* ALS  = XP + (size_t)NB * NN * HID;            // 65536
  float* ALD  = ALS + (size_t)NB * NN * NH;            // 65536
  unsigned* M = (unsigned*)(ALD + (size_t)NB * NN * NH); // 65536
  float* S    = (float*)(M + (size_t)NB * NN * NH);    // 65536
  float* AGG  = S + (size_t)NB * NN * NH;              // 524288
  float* H1   = AGG + (size_t)NB * NN * HID;           // 524288

  const int TB = 256;
  const int nRowBlk  = (NB * NN) / 128;                // 128
  const int nEdgeBlk = (NB * NE) / TB;                 // 2048
  const int nAggBlk  = (NB * NE * NH) / TB;            // 8192
  const int nElemBlk = (NB * NN * HID) / TB;           // 2048
  const int nRowThr  = (NB * NN) / TB;                 // 64

  // 1) stream-zero the 268MB attention output (dominant HBM cost)
  zero_a_kernel<<<8192, TB, 0, stream>>>((f4*)Amat,
      ((size_t)NB * NH * NN * NN) / 4);

  // 2) embedding MLP + LN (WMMA)
  embed_kernel<<<nRowBlk, TB, 0, stream>>>(Hin, W1e, b1e, W2e, b2e, lneg, lneb, X);

  // ---- GAT layer 1 ----
  init_ws_kernel<<<nElemBlk, TB, 0, stream>>>(M, S, AGG);
  xp_kernel<<<nRowBlk, TB, 0, stream>>>(X, Wg1, as1, ad1, XP, ALS, ALD);
  edge_max_kernel<<<nEdgeBlk, TB, 0, stream>>>(EI, ALS, ALD, M);
  edge_sum_kernel<<<nEdgeBlk, TB, 0, stream>>>(EI, ALS, ALD, M, S);
  aggregate_kernel<<<nAggBlk, TB, 0, stream>>>(EI, ALS, ALD, M, S, XP, AGG, Amat, 0);
  post1_kernel<<<nElemBlk, TB, 0, stream>>>(AGG, bi1, H1);

  // ---- GAT layer 2 ----
  init_ws_kernel<<<nElemBlk, TB, 0, stream>>>(M, S, AGG);
  xp_kernel<<<nRowBlk, TB, 0, stream>>>(H1, Wg2, as2, ad2, XP, ALS, ALD);
  edge_max_kernel<<<nEdgeBlk, TB, 0, stream>>>(EI, ALS, ALD, M);
  edge_sum_kernel<<<nEdgeBlk, TB, 0, stream>>>(EI, ALS, ALD, M, S);
  aggregate_kernel<<<nAggBlk, TB, 0, stream>>>(EI, ALS, ALD, M, S, XP, AGG, Amat, 1);
  post2_kernel<<<nRowThr, TB, 0, stream>>>(H1, AGG, bi2, lnog, lnob, Hout);
}
